// CoordGlobalIUVHeadAfterMaskBranch_90082644066514
// MI455X (gfx1250) — matrix operand order repaired and checked by reading, not compiled, not run on hardware
//
#include <hip/hip_runtime.h>

// ---------------------------------------------------------------------------
// CDNA5 (gfx1250) implementation of CoordGlobalIUVHeadAfterMaskBranch.
// Conv layers are implicit GEMMs on v_wmma_f32_16x16x32_bf16 (wave32),
// with 2-way N register blocking and async global->LDS staging of the
// input halo tile (GLOBAL_LOAD_ASYNC_TO_LDS_B128, tracked by ASYNCcnt).
// ---------------------------------------------------------------------------

typedef __attribute__((ext_vector_type(16))) __bf16 v16bf;
typedef __attribute__((ext_vector_type(8)))  float  v8f;

union Frag {
    uint4 u[2];   // 32 bytes
    v16bf v;
};

#if __has_builtin(__builtin_amdgcn_global_load_async_to_lds_b128) && \
    __has_builtin(__builtin_amdgcn_s_wait_asynccnt)
#define HAVE_ASYNC_LDS 1
#endif

#ifdef HAVE_ASYNC_LDS
// Exact parameter types per clang's diagnostic:
//   param0: int __vector(4) __device__ *   (global, non-const)
//   param1: int __vector(4) __shared__ *   (LDS)
typedef int a_v4i __attribute__((vector_size(16)));

__device__ __forceinline__ void lds_async_copy_b128(const void* g, void* l) {
    __builtin_amdgcn_global_load_async_to_lds_b128(
        (__attribute__((address_space(1))) a_v4i*)g,
        (__attribute__((address_space(3))) a_v4i*)l,
        0, 0);
}
#endif

__device__ __forceinline__ unsigned short f2bf(float f) {
    unsigned int u = __float_as_uint(f);
    unsigned int rounding = 0x7FFFu + ((u >> 16) & 1u);
    return (unsigned short)((u + rounding) >> 16);
}

// ---------------------------------------------------------------------------
// Kernel 1: rel_coord + NCHW->NHWC bf16 transpose, augmented to 160 channels.
//   act0[n][h][w][c]: c 0..127 = iuv_feats, 128 = rel_x, 129 = rel_y, 130..159 = 0
// ---------------------------------------------------------------------------
__global__ __launch_bounds__(256)
void prep_kernel(const float* __restrict__ s_logits,
                 const float* __restrict__ iuv,
                 const int*   __restrict__ im_inds,
                 const float* __restrict__ inst_loc,
                 const int*   __restrict__ fpn,
                 const int*   __restrict__ stride_p,
                 const float* __restrict__ soi,
                 unsigned short* __restrict__ act0,
                 int N, int C, int H, int W, int M)
{
    int idx = blockIdx.x * blockDim.x + threadIdx.x;
    int total = N * H * W;
    if (idx >= total) return;
    int w = idx % W;
    int h = (idx / W) % H;
    int n = idx / (W * H);

    int stride = stride_p[0];
    float locx = (float)(w * stride + (stride >> 1));
    float locy = (float)(h * stride + (stride >> 1));

    float sx = 0.f, sy = 0.f;
    int cnt = 0;
    for (int m = 0; m < M; ++m) {
        if (im_inds[m] == n) {
            float s   = s_logits[(size_t)m * H * W + (size_t)h * W + w];
            float inv = 1.0f / soi[fpn[m]];
            sx += (inst_loc[2 * m + 0] - locx) * inv * s;
            sy += (inst_loc[2 * m + 1] - locy) * inv * s;
            ++cnt;
        }
    }
    float rx = (cnt > 0) ? sx / (float)cnt : 0.f;
    float ry = (cnt > 0) ? sy / (float)cnt : 0.f;

    unsigned short* o = act0 + (size_t)idx * 160;
    const size_t hw = (size_t)h * W + w;
    for (int c = 0; c < 128; ++c)
        o[c] = f2bf(iuv[(((size_t)n * C + c) * H * W) + hw]);
    o[128] = f2bf(rx);
    o[129] = f2bf(ry);
    for (int c = 130; c < 160; ++c) o[c] = 0;
}

// ---------------------------------------------------------------------------
// Kernel 2: repack OIHW f32 weights into WMMA B-fragments (bf16).
//   wfrag[co_tile][tap][kb][lane][j]  (j = 0..15)
//   B layout: lane L holds column N = L%16, K = kb*32 + (L<16 ? 0 : 16) + j.
//   aug==1 maps augmented channels: k<128 -> ci k+2; k==128 -> ci 0; k==129 -> ci 1.
// ---------------------------------------------------------------------------
__global__ __launch_bounds__(256)
void repack_kernel(const float* __restrict__ wsrc,
                   unsigned short* __restrict__ wfrag,
                   int co_tiles, int taps, int cblk,
                   int cin_real, int co_real, int aug)
{
    int idx = blockIdx.x * blockDim.x + threadIdx.x;
    int total = co_tiles * taps * cblk * 32 * 16;
    if (idx >= total) return;
    int j    = idx % 16;
    int lane = (idx / 16) % 32;
    int kb   = (idx / (16 * 32)) % cblk;
    int tap  = (idx / (16 * 32 * cblk)) % taps;
    int cot  = idx / (16 * 32 * cblk * taps);
    int co   = cot * 16 + (lane & 15);
    int k    = kb * 32 + ((lane < 16) ? 0 : 16) + j;

    int wi;
    if (aug) wi = (k < 128) ? (k + 2) : ((k == 128) ? 0 : ((k == 129) ? 1 : -1));
    else     wi = (k < cin_real) ? k : -1;

    float val = 0.f;
    if (wi >= 0 && co < co_real)
        val = wsrc[((size_t)co * cin_real + wi) * taps + tap];
    wfrag[idx] = f2bf(val);
}

// ---------------------------------------------------------------------------
// Kernel 3: 3x3 conv + bias + ReLU via WMMA implicit GEMM.
//   Block = 256 threads (8 wave32). Output tile: 32 pixels x 128 channels.
//   Wave (mi = wave&1, npair = wave>>1): 16 pixels x 32 channels,
//   two accumulators sharing one A fragment (2x WMMA per A load).
//   LDS: input halo tile 3 rows x 34 cols x Cin bf16, staged with
//   GLOBAL_LOAD_ASYNC_TO_LDS_B128 when the builtin is available.
//   A k order used here (matches B repack): lane<16 holds M=lane, K in {0..15};
//                            lane>=16 holds M=lane-16, K in {16..31}.
// ---------------------------------------------------------------------------
template <int CBLK>
__global__ __launch_bounds__(256)
void conv3x3_wmma_kernel(const unsigned short* __restrict__ act_in,  // NHWC, Cin=CBLK*32
                         const unsigned short* __restrict__ wfrag,
                         const float* __restrict__ bias,
                         unsigned short* __restrict__ act_out,       // NHWC, 128
                         int H, int W)
{
    const int Cin = CBLK * 32;
    const int CH4 = CBLK * 4;                 // uint4 chunks per pixel
    const int TW  = 34;                       // halo tile width (32 + 2)
    __shared__ uint4 tile[3 * 34 * CBLK * 4]; // [3][34][Cin] bf16

    const int tid  = threadIdx.x;
    const int lane = tid & 31;
    const int wave = tid >> 5;
    const int w0   = blockIdx.x * 32;
    const int h    = blockIdx.y;
    const int n    = blockIdx.z;

    // Hint the weight fragments for this block into cache early.
    __builtin_prefetch(wfrag, 0, 3);

    // Cooperative halo load with zero padding at borders.
    const int TOT = 3 * TW * CH4;
    for (int idx = tid; idx < TOT; idx += 256) {
        int pix = idx / CH4, chunk = idx % CH4;
        int r = pix / TW, c = pix % TW;
        int gh = h - 1 + r, gw = w0 - 1 + c;
        if (gh >= 0 && gh < H && gw >= 0 && gw < W) {
            const uint4* g = (const uint4*)(act_in +
                ((size_t)((size_t)n * H + gh) * W + gw) * Cin) + chunk;
#ifdef HAVE_ASYNC_LDS
            lds_async_copy_b128(g, &tile[idx]);
#else
            tile[idx] = *g;
#endif
        } else {
            tile[idx] = make_uint4(0u, 0u, 0u, 0u);
        }
    }
#ifdef HAVE_ASYNC_LDS
    __builtin_amdgcn_s_wait_asynccnt(0);
#endif
    __syncthreads();

    v8f acc0 = {0.f, 0.f, 0.f, 0.f, 0.f, 0.f, 0.f, 0.f};
    v8f acc1 = {0.f, 0.f, 0.f, 0.f, 0.f, 0.f, 0.f, 0.f};
    const int mrow   = lane & 15;
    const int hiHalf = lane >> 4;
    const int mi     = wave & 1;              // which 16-pixel half
    const int npair  = wave >> 1;             // which pair of co-tiles
    const int cot0   = npair * 2;
    const int cot1   = npair * 2 + 1;

#pragma unroll
    for (int tap = 0; tap < 9; ++tap) {
        const int dh = tap / 3, dw = tap % 3;           // offsets 0..2 (halo-relative)
        const int pixBase = (dh * TW + mi * 16 + mrow + dw) * CH4;
#pragma unroll
        for (int kb = 0; kb < CBLK; ++kb) {
            Frag A, B0, B1;
            A.u[0] = tile[pixBase + kb * 4 + 2 * hiHalf];
            A.u[1] = tile[pixBase + kb * 4 + 2 * hiHalf + 1];
            const uint4* wp0 = (const uint4*)wfrag +
                ((size_t)(((size_t)cot0 * 9 + tap) * CBLK + kb) * 32 + lane) * 2;
            const uint4* wp1 = (const uint4*)wfrag +
                ((size_t)(((size_t)cot1 * 9 + tap) * CBLK + kb) * 32 + lane) * 2;
            B0.u[0] = wp0[0]; B0.u[1] = wp0[1];
            B1.u[0] = wp1[0]; B1.u[1] = wp1[1];
            acc0 = __builtin_amdgcn_wmma_f32_16x16x32_bf16(
                false, A.v, false, B0.v, (short)0, acc0, false, false);
            acc1 = __builtin_amdgcn_wmma_f32_16x16x32_bf16(
                false, A.v, false, B1.v, (short)0, acc1, false, false);
        }
    }

    // Epilogue: bias + ReLU + bf16 store. C/D layout: N=lane%16, M=r+8*hiHalf.
#pragma unroll
    for (int t = 0; t < 2; ++t) {
        const int co = (npair * 2 + t) * 16 + mrow;
        const float b = bias[co];
        const v8f acc = t ? acc1 : acc0;
#pragma unroll
        for (int r = 0; r < 8; ++r) {
            int m = r + hiHalf * 8;
            float v = acc[r] + b;
            v = v > 0.f ? v : 0.f;
            act_out[((size_t)((size_t)n * H + h) * W + (w0 + mi * 16 + m)) * 128 + co]
                = f2bf(v);
        }
    }
}

// ---------------------------------------------------------------------------
// Kernel 4: 1x1 conv (128 -> 75, padded to 80) via WMMA. f32 NCHW output.
//   Block = 160 threads (5 wave32), 16 pixels per block.
// ---------------------------------------------------------------------------
__global__ __launch_bounds__(160)
void conv1x1_wmma_kernel(const unsigned short* __restrict__ act_in,  // NHWC, 128
                         const unsigned short* __restrict__ wfrag,
                         const float* __restrict__ bias,
                         float* __restrict__ out,                    // NCHW, OUT
                         int H, int W, int OUT)
{
    __shared__ uint4 tile[16 * 16];           // 16 pixels x 128 bf16
    const int tid  = threadIdx.x;
    const int lane = tid & 31;
    const int wave = tid >> 5;
    const int w0   = blockIdx.x * 16;
    const int h    = blockIdx.y;
    const int n    = blockIdx.z;

    for (int idx = tid; idx < 256; idx += 160) {
        int pix = idx >> 4, chunk = idx & 15;
        const uint4* g = ((const uint4*)(act_in +
            ((size_t)((size_t)n * H + h) * W + (w0 + pix)) * 128)) + chunk;
#ifdef HAVE_ASYNC_LDS
        lds_async_copy_b128(g, &tile[idx]);
#else
        tile[idx] = *g;
#endif
    }
#ifdef HAVE_ASYNC_LDS
    __builtin_amdgcn_s_wait_asynccnt(0);
#endif
    __syncthreads();

    v8f acc = {0.f, 0.f, 0.f, 0.f, 0.f, 0.f, 0.f, 0.f};
    const int mrow   = lane & 15;
    const int hiHalf = lane >> 4;

#pragma unroll
    for (int kb = 0; kb < 4; ++kb) {
        Frag A, B;
        const int pixBase = mrow * 16;
        A.u[0] = tile[pixBase + kb * 4 + 2 * hiHalf];
        A.u[1] = tile[pixBase + kb * 4 + 2 * hiHalf + 1];
        const uint4* wp = (const uint4*)wfrag + ((size_t)(wave * 4 + kb) * 32 + lane) * 2;
        B.u[0] = wp[0];
        B.u[1] = wp[1];
        acc = __builtin_amdgcn_wmma_f32_16x16x32_bf16(
            false, A.v, false, B.v, (short)0, acc, false, false);
    }

    const int co = wave * 16 + mrow;
    if (co < OUT) {
        const float b = bias[co];
#pragma unroll
        for (int r = 0; r < 8; ++r) {
            int m = r + hiHalf * 8;
            out[(((size_t)n * OUT + co) * H + h) * W + (w0 + m)] = acc[r] + b;
        }
    }
}

// ---------------------------------------------------------------------------
// Kernel 5: aligned_bilinear(x, 2) closed form.
//   out[oy][ox]: iy=max(oy-1,0), ix=max(ox-1,0);
//   even index -> copy, odd -> 0.5 blend of neighbors (edge-clamped).
// ---------------------------------------------------------------------------
__global__ __launch_bounds__(256)
void upsample2_kernel(const float* __restrict__ in, float* __restrict__ out,
                      long long total, int H, int W)
{
    long long idx = (long long)blockIdx.x * blockDim.x + threadIdx.x;
    if (idx >= total) return;
    const int W2 = 2 * W, H2 = 2 * H;
    int ox = (int)(idx % W2);
    int oy = (int)((idx / W2) % H2);
    long long nc = idx / ((long long)W2 * H2);

    int iy = oy > 0 ? oy - 1 : 0;
    int ix = ox > 0 ? ox - 1 : 0;
    int y0 = iy >> 1, x0 = ix >> 1;
    int fy = iy & 1, fx = ix & 1;
    int y1 = y0 + 1 < H ? y0 + 1 : H - 1;
    int x1 = x0 + 1 < W ? x0 + 1 : W - 1;

    const float* p = in + nc * (long long)H * W;
    float v;
    if (!fy && !fx)      v = p[(size_t)y0 * W + x0];
    else if (!fy)        v = 0.5f * (p[(size_t)y0 * W + x0] + p[(size_t)y0 * W + x1]);
    else if (!fx)        v = 0.5f * (p[(size_t)y0 * W + x0] + p[(size_t)y1 * W + x0]);
    else                 v = 0.25f * (p[(size_t)y0 * W + x0] + p[(size_t)y0 * W + x1] +
                                      p[(size_t)y1 * W + x0] + p[(size_t)y1 * W + x1]);
    out[idx] = v;
}

// ---------------------------------------------------------------------------
// Host launcher.
// ---------------------------------------------------------------------------
extern "C" void kernel_launch(void* const* d_in, const int* in_sizes, int n_in,
                              void* d_out, int out_size, void* d_ws, size_t ws_size,
                              hipStream_t stream)
{
    (void)n_in; (void)out_size; (void)ws_size;

    const float* s_logits = (const float*)d_in[0];
    const float* iuv      = (const float*)d_in[1];
    const int*   im_inds  = (const int*)  d_in[2];
    const float* inst_loc = (const float*)d_in[3];
    const int*   fpn      = (const int*)  d_in[4];
    const int*   stride_p = (const int*)  d_in[5];
    const float* soi      = (const float*)d_in[6];
    const float* w0 = (const float*)d_in[7];   const float* b0 = (const float*)d_in[8];
    const float* w1 = (const float*)d_in[9];   const float* b1 = (const float*)d_in[10];
    const float* w2 = (const float*)d_in[11];  const float* b2 = (const float*)d_in[12];
    const float* w3 = (const float*)d_in[13];  const float* b3 = (const float*)d_in[14];
    const float* wf = (const float*)d_in[15];  const float* bf = (const float*)d_in[16];

    const int N = 4, C = 128, H = 160, W = 160, OUT = 75;
    const int M = in_sizes[2];                     // im_inds length

    char* ws = (char*)d_ws;
    size_t off = 0;
    auto take = [&](size_t bytes) { size_t o = off; off += (bytes + 255) & ~(size_t)255; return o; };

    const size_t off_act0 = take((size_t)N * H * W * 160 * 2);   // augmented bf16 NHWC
    const size_t off_actA = take((size_t)N * H * W * 128 * 2);
    const size_t off_actB = take((size_t)N * H * W * 128 * 2);
    const size_t off_wf0  = take((size_t)8 * 9 * 5 * 32 * 16 * 2);
    const size_t off_wf1  = take((size_t)8 * 9 * 4 * 32 * 16 * 2);
    const size_t off_wf2  = take((size_t)8 * 9 * 4 * 32 * 16 * 2);
    const size_t off_wf3  = take((size_t)8 * 9 * 4 * 32 * 16 * 2);
    const size_t off_wff  = take((size_t)5 * 1 * 4 * 32 * 16 * 2);
    const size_t off_conv = off_act0;              // act0 dead after layer 0; alias it

    unsigned short* act0 = (unsigned short*)(ws + off_act0);
    unsigned short* actA = (unsigned short*)(ws + off_actA);
    unsigned short* actB = (unsigned short*)(ws + off_actB);
    unsigned short* wf0  = (unsigned short*)(ws + off_wf0);
    unsigned short* wf1  = (unsigned short*)(ws + off_wf1);
    unsigned short* wf2  = (unsigned short*)(ws + off_wf2);
    unsigned short* wf3  = (unsigned short*)(ws + off_wf3);
    unsigned short* wff  = (unsigned short*)(ws + off_wff);
    float* convout       = (float*)(ws + off_conv);

    // 1) rel_coord + augmented NHWC bf16 activation
    {
        int total = N * H * W;
        prep_kernel<<<(total + 255) / 256, 256, 0, stream>>>(
            s_logits, iuv, im_inds, inst_loc, fpn, stride_p, soi,
            act0, N, C, H, W, M);
    }

    // 2) weight repacks
    {
        int t0 = 8 * 9 * 5 * 32 * 16;
        repack_kernel<<<(t0 + 255) / 256, 256, 0, stream>>>(w0, wf0, 8, 9, 5, 130, 128, 1);
        int t1 = 8 * 9 * 4 * 32 * 16;
        repack_kernel<<<(t1 + 255) / 256, 256, 0, stream>>>(w1, wf1, 8, 9, 4, 128, 128, 0);
        repack_kernel<<<(t1 + 255) / 256, 256, 0, stream>>>(w2, wf2, 8, 9, 4, 128, 128, 0);
        repack_kernel<<<(t1 + 255) / 256, 256, 0, stream>>>(w3, wf3, 8, 9, 4, 128, 128, 0);
        int tf = 5 * 1 * 4 * 32 * 16;
        repack_kernel<<<(tf + 255) / 256, 256, 0, stream>>>(wf, wff, 5, 1, 4, 128, OUT, 0);
    }

    // 3) four 3x3 conv + ReLU layers (WMMA implicit GEMM, 32px x 128ch blocks)
    dim3 cgrid(W / 32, H, N);
    conv3x3_wmma_kernel<5><<<cgrid, 256, 0, stream>>>(act0, wf0, b0, actA, H, W);
    conv3x3_wmma_kernel<4><<<cgrid, 256, 0, stream>>>(actA, wf1, b1, actB, H, W);
    conv3x3_wmma_kernel<4><<<cgrid, 256, 0, stream>>>(actB, wf2, b2, actA, H, W);
    conv3x3_wmma_kernel<4><<<cgrid, 256, 0, stream>>>(actA, wf3, b3, actB, H, W);

    // 4) 1x1 conv to 75 channels (WMMA), f32 NCHW
    dim3 fgrid(W / 16, H, N);
    conv1x1_wmma_kernel<<<fgrid, 160, 0, stream>>>(actB, wff, bf, convout, H, W, OUT);

    // 5) aligned_bilinear x2 -> d_out (4, 75, 320, 320)
    {
        long long total = (long long)N * OUT * (2 * H) * (2 * W);
        upsample2_kernel<<<(unsigned)((total + 255) / 256), 256, 0, stream>>>(
            convout, (float*)d_out, total, H, W);
    }
}
